// Attention_8220567404931
// MI455X (gfx1250) — matrix operand order repaired and checked by reading, not compiled, not run on hardware
//
#include <hip/hip_runtime.h>
#include <hip/hip_bf16.h>
#include <math.h>

// ---------- WMMA fragment types (wave32, CDNA5) ----------
typedef __attribute__((ext_vector_type(16))) __bf16 v16bf;
typedef __attribute__((ext_vector_type(8)))  __bf16 v8bf;
typedef __attribute__((ext_vector_type(8)))  float  v8f;

#define DIMX     1024
#define DIM_HEAD 64
#define HEADS    8
#define INNER    512          // HEADS*DIM_HEAD
#define KVDIM    128          // 2*DIM_HEAD
#define NCOLS    640          // INNER + KVDIM
#define SEQ      2048
#define BATCH    4
#define ROWS     (BATCH*SEQ)  // 8192
#define QK_SCALE 0.125f       // 64^-0.5

// ---------- async global->LDS copy (CDNA5, ASYNCcnt) ----------
// Builtin signature (from compiler diagnostic): param1 = global ptr to
// 'int __attribute__((vector_size(16)))', param2 = LDS ptr, then imm offset, imm cpol.
typedef int v4i_g __attribute__((vector_size(16)));
typedef __attribute__((address_space(1))) v4i_g* gb128_ptr;
typedef __attribute__((address_space(3))) v4i_g* lb128_ptr;

__device__ __forceinline__ void async_copy16(const void* g, void* l) {
#if __has_builtin(__builtin_amdgcn_global_load_async_to_lds_b128)
    __builtin_amdgcn_global_load_async_to_lds_b128(
        (gb128_ptr)(unsigned long long)(size_t)g,   // global src (flat value == global value)
        (lb128_ptr)(unsigned)(size_t)l,             // LDS offset = low 32 bits of generic ptr
        0, 0);
#else
    *(v8bf*)l = *(const v8bf*)g;                    // sync fallback: load + ds_store
#endif
}

__device__ __forceinline__ void wait_async0() {
#if __has_builtin(__builtin_amdgcn_global_load_async_to_lds_b128)
#if __has_builtin(__builtin_amdgcn_s_wait_asynccnt)
    __builtin_amdgcn_s_wait_asynccnt(0);
#else
    asm volatile("s_wait_asynccnt 0" ::: "memory");
#endif
#else
    asm volatile("s_wait_dscnt 0" ::: "memory");
#endif
}

// ---------- fragment loaders ----------
// A fragment 16x32 bf16, row-major source. base points at tile origin (row0,k0).
// Lane l: row m = l&15; elements 0..7 hold K = 8*hi+0..7, elements 8..15 hold K = 16+8*hi+0..7.
__device__ __forceinline__ v16bf load_a_frag(const __bf16* base, int ld, int lane) {
    int m = lane & 15, hi = lane >> 4;
    const __bf16* p = base + (size_t)m * ld + hi * 8;
    union { v16bf v; v8bf h[2]; } u;
    u.h[0] = *(const v8bf*)(p);
    u.h[1] = *(const v8bf*)(p + 16);
    return u.v;
}

// B fragment 32x16 bf16 from TRANSPOSED B (Bt row = output column; elements 0..7 hold K = 8*kh+0..7, 8..15 hold K = 16+8*kh+0..7).
__device__ __forceinline__ v16bf load_b_frag(const __bf16* baseT, int ldt, int lane) {
    int n = lane & 15, kh = lane >> 4;
    const __bf16* p = baseT + (size_t)n * ldt + kh * 8;
    union { v16bf v; v8bf h[2]; } u;
    u.h[0] = *(const v8bf*)(p);
    u.h[1] = *(const v8bf*)(p + 16);
    return u.v;
}

__device__ __forceinline__ v8f wmma_bf16(v16bf a, v16bf b, v8f c) {
    return __builtin_amdgcn_wmma_f32_16x16x32_bf16(
        /*neg_a=*/false, a, /*neg_b=*/false, b,
        /*c_mod=*/(short)0, c, /*reuse_a=*/false, /*reuse_b=*/false);
}

// ---------- kernel 1: weights -> bf16 transposed ----------
__global__ void prep_weights_k(const float* __restrict__ Wq,
                               const float* __restrict__ Wkv,
                               const float* __restrict__ Wo,
                               __bf16* __restrict__ WcT,   // [640][1024]
                               __bf16* __restrict__ WoT) { // [1024][512]
    int i = blockIdx.x * blockDim.x + threadIdx.x;
    const int nc = NCOLS * DIMX;
    if (i < nc) {
        int c = i / DIMX, k = i % DIMX;
        float v = (c < INNER) ? Wq[(size_t)k * INNER + c]
                              : Wkv[(size_t)k * KVDIM + (c - INNER)];
        WcT[(size_t)c * DIMX + k] = (__bf16)v;
    } else if (i < nc + DIMX * INNER) {
        int j = i - nc;
        int c = j / INNER, k = j % INNER;
        WoT[(size_t)c * INNER + k] = (__bf16)Wo[(size_t)k * DIMX + c];
    }
}

// ---------- kernel 2/6: row layernorm (DIM=1024 per row, 256 threads) ----------
template <typename OUT>
__global__ void ln_k(const float* __restrict__ x, const float* __restrict__ w,
                     const float* __restrict__ b, OUT* __restrict__ y) {
    __shared__ float red0[8], red1[8];
    int row = blockIdx.x;
    const float* xr = x + (size_t)row * DIMX;
    float s = 0.f, s2 = 0.f;
    for (int i = threadIdx.x; i < DIMX; i += 256) {
        float v = xr[i]; s += v; s2 += v * v;
    }
    #pragma unroll
    for (int m = 16; m >= 1; m >>= 1) {
        s  += __shfl_xor(s,  m, 32);
        s2 += __shfl_xor(s2, m, 32);
    }
    int wave = threadIdx.x >> 5;
    if ((threadIdx.x & 31) == 0) { red0[wave] = s; red1[wave] = s2; }
    __syncthreads();
    float ts = 0.f, ts2 = 0.f;
    #pragma unroll
    for (int i = 0; i < 8; i++) { ts += red0[i]; ts2 += red1[i]; }
    float mu  = ts / (float)DIMX;
    float var = ts2 / (float)DIMX - mu * mu;
    float rs  = rsqrtf(var + 1e-5f);
    OUT* yr = y + (size_t)row * DIMX;
    for (int i = threadIdx.x; i < DIMX; i += 256)
        yr[i] = (OUT)((xr[i] - mu) * rs * w[i] + b[i]);
}

// ---------- kernel 3: fused QKV projection (8192x1024 @ 1024x640, bf16 WMMA) ----------
// 8 waves share one 16-row A panel -> stage A in LDS via async copies (8x traffic cut).
__global__ void gemm_qkv_k(const __bf16* __restrict__ xn,   // [8192][1024]
                           const __bf16* __restrict__ WcT,  // [640][1024]
                           __bf16* __restrict__ qb,         // [b][h][n][64] (pre-scaled)
                           __bf16* __restrict__ kb,         // [b][n][64]
                           __bf16* __restrict__ vT) {       // [b][64][n]
    __shared__ __bf16 atile[16 * 128];   // 4KB A chunk: 16 rows x 128 k
    int lane = threadIdx.x & 31, wave = threadIdx.x >> 5;
    int rt = blockIdx.x;                 // 0..511
    int ct = blockIdx.y * 8 + wave;      // 0..39
    int row0 = rt * 16, c0 = ct * 16;
    const __bf16* arow = xn  + (size_t)row0 * DIMX;
    const __bf16* brow = WcT + (size_t)c0   * DIMX;
    int ar = threadIdx.x >> 4, ac = (threadIdx.x & 15) * 8;  // 16 rows x 16 chunks of 16B
    v8f acc = {};
    for (int ks = 0; ks < DIMX; ks += 128) {
        __syncthreads();
        async_copy16(arow + (size_t)ar * DIMX + ks + ac, &atile[ar * 128 + ac]);
        wait_async0();
        __syncthreads();
        #pragma unroll
        for (int kk = 0; kk < 128; kk += 32) {
            __builtin_prefetch(brow + ks + kk + 256, 0, 0);
            v16bf a = load_a_frag(atile + kk, 128, lane);
            v16bf b = load_b_frag(brow + ks + kk, DIMX, lane);
            acc = wmma_bf16(a, b, acc);
        }
    }
    // epilogue: route columns to Q / K / V^T
    int n = lane & 15, hi = lane >> 4;
    int c = c0 + n;
    #pragma unroll
    for (int r = 0; r < 8; r++) {
        int grow = row0 + hi * 8 + r;
        int bidx = grow / SEQ, srow = grow % SEQ;
        float v = acc[r];
        if (c < INNER) {
            int h = c / DIM_HEAD, d = c % DIM_HEAD;
            qb[(((size_t)bidx * HEADS + h) * SEQ + srow) * DIM_HEAD + d] =
                (__bf16)(v * QK_SCALE);
        } else if (c < INNER + DIM_HEAD) {
            kb[((size_t)bidx * SEQ + srow) * DIM_HEAD + (c - INNER)] = (__bf16)v;
        } else {
            vT[((size_t)bidx * DIM_HEAD + (c - INNER - DIM_HEAD)) * SEQ + srow] = (__bf16)v;
        }
    }
}

// ---------- kernel 4: flash attention ----------
// 8 waves of a block handle 8 consecutive q-tiles of the SAME (b,h) -> they walk the
// same K/V blocks in lockstep. Stage each 32-key K tile (32x64) and V^T tile (64x32)
// in LDS via async copies, shared by all 8 waves (8x K/V traffic cut).
__global__ void flash_attn_k(const __bf16* __restrict__ qb,
                             const __bf16* __restrict__ kb,
                             const __bf16* __restrict__ vT,
                             __bf16* __restrict__ ob) {      // [b][n][512]
    __shared__ __bf16 ktile[32 * 64];      // 4KB: K rows j0..j0+31 (row-major, ld=64)
    __shared__ __bf16 vtile[64 * 32];      // 4KB: V^T rows d=0..63, keys j0..j0+31 (ld=32)
    __shared__ __bf16 ptile[8][16 * 32];   // per-wave P staging (C-layout -> A-layout)
    int lane = threadIdx.x & 31, wave = threadIdx.x >> 5;
    int task = blockIdx.x * 8 + wave;      // 0..4095 ; 8|128 => whole block shares (b,h)
    int qt   = task & 127;
    int h    = (task >> 7) & 7;
    int bidx = task >> 10;

    const __bf16* qbase = qb + (((size_t)bidx * HEADS + h) * SEQ + qt * 16) * DIM_HEAD;
    v16bf aq0 = load_a_frag(qbase,      DIM_HEAD, lane);
    v16bf aq1 = load_a_frag(qbase + 32, DIM_HEAD, lane);
    const __bf16* kbase = kb + (size_t)bidx * SEQ * DIM_HEAD;
    const __bf16* vbase = vT + (size_t)bidx * DIM_HEAD * SEQ;

    int n = lane & 15, hi = lane >> 4;
    float mrow[8], lrow[8];
    #pragma unroll
    for (int r = 0; r < 8; r++) { mrow[r] = -1e30f; lrow[r] = 0.f; }
    v8f o0 = {}, o1 = {}, o2 = {}, o3 = {};
    __bf16* myp = &ptile[wave][0];

    // per-thread copy slots: K = 32 rows x 8 chunks(16B); V = 64 rows x 4 chunks(16B)
    int kr = threadIdx.x >> 3, kc = (threadIdx.x & 7) * 8;
    int vr = threadIdx.x >> 2, vc = (threadIdx.x & 3) * 8;

    for (int j0 = 0; j0 < SEQ; j0 += 32) {
        __syncthreads();   // previous iteration's LDS reads complete
        async_copy16(kbase + (size_t)(j0 + kr) * DIM_HEAD + kc, &ktile[kr * 64 + kc]);
        async_copy16(vbase + (size_t)vr * SEQ + j0 + vc,        &vtile[vr * 32 + vc]);
        wait_async0();
        __syncthreads();   // tiles visible to all waves

        // S = Q @ K^T : two 16x16 key tiles, contraction d=64 (2 WMMA k-steps each)
        v8f s0 = {}, s1 = {};
        s0 = wmma_bf16(aq0, load_b_frag(ktile,            64, lane), s0);
        s0 = wmma_bf16(aq1, load_b_frag(ktile + 32,       64, lane), s0);
        s1 = wmma_bf16(aq0, load_b_frag(ktile + 16 * 64,      64, lane), s1);
        s1 = wmma_bf16(aq1, load_b_frag(ktile + 16 * 64 + 32, 64, lane), s1);

        // online softmax: C layout -> lane holds col n, rows 8*hi+r; reduce across 16-lane half
        #pragma unroll
        for (int r = 0; r < 8; r++) {
            float tm = fmaxf(s0[r], s1[r]);
            tm = fmaxf(tm, __shfl_xor(tm, 1, 32));
            tm = fmaxf(tm, __shfl_xor(tm, 2, 32));
            tm = fmaxf(tm, __shfl_xor(tm, 4, 32));
            tm = fmaxf(tm, __shfl_xor(tm, 8, 32));
            float mn = fmaxf(mrow[r], tm);
            float sc = __expf(mrow[r] - mn);
            float p0 = __expf(s0[r] - mn);
            float p1 = __expf(s1[r] - mn);
            float ps = p0 + p1;
            ps += __shfl_xor(ps, 1, 32);
            ps += __shfl_xor(ps, 2, 32);
            ps += __shfl_xor(ps, 4, 32);
            ps += __shfl_xor(ps, 8, 32);
            lrow[r] = lrow[r] * sc + ps;
            mrow[r] = mn;
            o0[r] *= sc; o1[r] *= sc; o2[r] *= sc; o3[r] *= sc;
            myp[(hi * 8 + r) * 32 + n]      = (__bf16)p0;
            myp[(hi * 8 + r) * 32 + n + 16] = (__bf16)p1;
        }
        asm volatile("s_wait_dscnt 0" ::: "memory");  // LDS write -> read, same wave
        v16bf ap = load_a_frag(myp, 32, lane);
        // O += P @ V from LDS V^T tile (ld=32)
        o0 = wmma_bf16(ap, load_b_frag(vtile,           32, lane), o0);
        o1 = wmma_bf16(ap, load_b_frag(vtile + 16 * 32, 32, lane), o1);
        o2 = wmma_bf16(ap, load_b_frag(vtile + 32 * 32, 32, lane), o2);
        o3 = wmma_bf16(ap, load_b_frag(vtile + 48 * 32, 32, lane), o3);
    }

    #pragma unroll
    for (int r = 0; r < 8; r++) {
        float inv = 1.0f / lrow[r];
        int srow = qt * 16 + hi * 8 + r;
        __bf16* dst = ob + ((size_t)bidx * SEQ + srow) * INNER + h * DIM_HEAD;
        dst[n]      = (__bf16)(o0[r] * inv);
        dst[n + 16] = (__bf16)(o1[r] * inv);
        dst[n + 32] = (__bf16)(o2[r] * inv);
        dst[n + 48] = (__bf16)(o3[r] * inv);
    }
}

// ---------- kernel 5: output projection (8192x512 @ 512x1024) ----------
__global__ void gemm_out_k(const __bf16* __restrict__ ob,   // [8192][512]
                           const __bf16* __restrict__ WoT,  // [1024][512]
                           float* __restrict__ out2) {      // [8192][1024]
    __shared__ __bf16 atile[16 * 128];
    int lane = threadIdx.x & 31, wave = threadIdx.x >> 5;
    int rt = blockIdx.x;                 // 0..511
    int ct = blockIdx.y * 8 + wave;      // 0..63
    int row0 = rt * 16, c0 = ct * 16;
    const __bf16* arow = ob  + (size_t)row0 * INNER;
    const __bf16* brow = WoT + (size_t)c0   * INNER;
    int ar = threadIdx.x >> 4, ac = (threadIdx.x & 15) * 8;
    v8f acc = {};
    for (int ks = 0; ks < INNER; ks += 128) {
        __syncthreads();
        async_copy16(arow + (size_t)ar * INNER + ks + ac, &atile[ar * 128 + ac]);
        wait_async0();
        __syncthreads();
        #pragma unroll
        for (int kk = 0; kk < 128; kk += 32) {
            __builtin_prefetch(brow + ks + kk + 256, 0, 0);
            acc = wmma_bf16(load_a_frag(atile + kk, 128, lane),
                            load_b_frag(brow + ks + kk, INNER, lane), acc);
        }
    }
    int n = lane & 15, hi = lane >> 4;
    #pragma unroll
    for (int r = 0; r < 8; r++)
        out2[(size_t)(row0 + hi * 8 + r) * DIMX + c0 + n] = acc[r];
}

// ---------- launcher ----------
extern "C" void kernel_launch(void* const* d_in, const int* in_sizes, int n_in,
                              void* d_out, int out_size, void* d_ws, size_t ws_size,
                              hipStream_t stream) {
    const float* x          = (const float*)d_in[0];
    const float* Wq         = (const float*)d_in[1];
    const float* Wkv        = (const float*)d_in[2];
    const float* Wo         = (const float*)d_in[3];
    const float* norm_w     = (const float*)d_in[4];
    const float* norm_b     = (const float*)d_in[5];
    const float* out_norm_w = (const float*)d_in[6];
    const float* out_norm_b = (const float*)d_in[7];
    float* out = (float*)d_out;

    unsigned char* ws = (unsigned char*)d_ws;
    size_t off = 0;
    __bf16* xn   = (__bf16*)(ws + off); off += (size_t)ROWS * DIMX * 2;                    // 16 MB
    __bf16* qb   = (__bf16*)(ws + off); off += (size_t)BATCH * HEADS * SEQ * DIM_HEAD * 2; // 8 MB
    __bf16* kb   = (__bf16*)(ws + off); off += (size_t)BATCH * SEQ * DIM_HEAD * 2;         // 1 MB
    __bf16* vT   = (__bf16*)(ws + off); off += (size_t)BATCH * DIM_HEAD * SEQ * 2;         // 1 MB
    __bf16* ob   = (__bf16*)(ws + off); off += (size_t)ROWS * INNER * 2;                   // 8 MB
    __bf16* WcT  = (__bf16*)(ws + off); off += (size_t)NCOLS * DIMX * 2;                   // 1.25 MB
    __bf16* WoT  = (__bf16*)(ws + off); off += (size_t)DIMX * INNER * 2;                   // 1 MB
    float*  out2 = (float*) (ws + off);                                                    // 32 MB

    {
        int total = NCOLS * DIMX + DIMX * INNER;
        prep_weights_k<<<(total + 255) / 256, 256, 0, stream>>>(Wq, Wkv, Wo, WcT, WoT);
    }
    ln_k<__bf16><<<ROWS, 256, 0, stream>>>(x, norm_w, norm_b, xn);
    gemm_qkv_k<<<dim3(ROWS / 16, 5), 256, 0, stream>>>(xn, WcT, qb, kb, vT);
    flash_attn_k<<<(BATCH * HEADS * (SEQ / 16)) / 8, 256, 0, stream>>>(qb, kb, vT, ob);
    gemm_out_k<<<dim3(ROWS / 16, 8), 256, 0, stream>>>(ob, WoT, out2);
    ln_k<float><<<ROWS, 256, 0, stream>>>(out2, out_norm_w, out_norm_b, out);
}
